// HBondGNNEncoder_23012434772712
// MI455X (gfx1250) — matrix operand order repaired and checked by reading, not compiled, not run on hardware
//
#include <hip/hip_runtime.h>
#include <hip/hip_bf16.h>

#define STR 136          // padded LDS row stride in bf16 elements (128 + 8)
#define NTHREADS 160     // 5 wave32s; 8 batches * 20 nodes = 160 rows
#define BPW 8            // batches per workgroup
#define ROWS 160         // BPW * 20
#define HID 128

typedef __attribute__((ext_vector_type(16))) __bf16 v16bf;
typedef __attribute__((ext_vector_type(8)))  __bf16 v8bf;
typedef __attribute__((ext_vector_type(8)))  float  v8f;
typedef __attribute__((ext_vector_type(4)))  unsigned int u32x4;
typedef __attribute__((ext_vector_type(8)))  int          i32x8;
typedef __attribute__((ext_vector_type(4)))  int          i32x4;

__device__ __forceinline__ v16bf load_frag(const __bf16* p0, const __bf16* p1) {
    v8bf a = *(const v8bf*)p0;
    v8bf b = *(const v8bf*)p1;
    return __builtin_shufflevector(a, b, 0,1,2,3,4,5,6,7,8,9,10,11,12,13,14,15);
}

__device__ __forceinline__ float gelu_exact(float x) {
    return 0.5f * x * (1.0f + erff(x * 0.70710678118654752f));
}

// Tensor Data Mover: 1-D contiguous copy of `ndw` dwords global -> LDS.
// D# group0: count=1 | lds_addr | global_addr | type=2 ("image").
// D# group1: data_size=4B; tensor_dim0 = tile_dim0 = ndw; tensor_dim1=1.
// This toolchain exposes the 6-arg builtin: (g0, g1, g2, g3, g4, cpol).
__device__ __forceinline__ void tdm_copy_dwords(const void* gsrc,
                                                unsigned lds_off,
                                                unsigned ndw) {
    unsigned long long ga = (unsigned long long)gsrc;
    u32x4 g0;
    g0[0] = 1u;                                           // count=1 (valid), user mode
    g0[1] = lds_off;                                      // lds_addr (bytes)
    g0[2] = (unsigned)(ga & 0xFFFFFFFFu);                 // global_addr[31:0]
    g0[3] = (unsigned)((ga >> 32) & 0x01FFFFFFu)          // global_addr[56:32]
          | (2u << 30);                                   // type=2
    i32x8 g1;
    g1[0] = 0x20000;                                      // data_size=2 -> 4 bytes
    g1[1] = (int)((ndw & 0xFFFFu) << 16);                 // tensor_dim0[15:0]
    g1[2] = (int)(((ndw >> 16) & 0xFFFFu) | (1u << 16));  // dim0[31:16], tensor_dim1=1
    g1[3] = (int)((ndw & 0xFFFFu) << 16);                 // tile_dim0 = ndw
    g1[4] = 0;                                            // tile_dim1/2 unused
    g1[5] = (int)ndw;                                     // tensor_dim0_stride
    g1[6] = 0;
    g1[7] = 0;
    i32x4 z4 = {0, 0, 0, 0};
    i32x8 z8 = {0, 0, 0, 0, 0, 0, 0, 0};
    __builtin_amdgcn_tensor_load_to_lds(g0, g1, z4, z4, z8, 0);
}

// D = A(16x128 rows of tile t) @ W^T (staged transposed: sW[col*STR + k]) + bias,
// then layernorm across the 128-wide row.  out[nt][v] holds col = nt*16+(lane&15),
// row-in-tile = v + (lane>>4)*8, matching the 16x16 f32 C/D VGPR layout.
__device__ __forceinline__ void mm_ln(const __bf16* __restrict__ sA,
                                      const __bf16* __restrict__ sW,
                                      const float*  __restrict__ bias,
                                      const float*  __restrict__ g,
                                      const float*  __restrict__ be,
                                      int t, int lane, float out[8][8]) {
    const int r = lane & 15;
    const int half = lane >> 4;
    const int row = t * 16 + r;

    // A fragments for all 4 K-steps (16x32 bf16 each)
    v16bf afr[4];
#pragma unroll
    for (int kk = 0; kk < 4; ++kk) {
        const __bf16* p = &sA[row * STR + kk * 32 + half * 8];
        afr[kk] = load_frag(p, p + 16);
    }

#pragma unroll
    for (int nt = 0; nt < 8; ++nt) {
        const int col = nt * 16 + r;
        v8f acc = {};
#pragma unroll
        for (int kk = 0; kk < 4; ++kk) {
            const __bf16* q = &sW[col * STR + kk * 32 + half * 8];
            v16bf bfr = load_frag(q, q + 16);
            acc = __builtin_amdgcn_wmma_f32_16x16x32_bf16(
                false, afr[kk], false, bfr, (short)0, acc, false, false);
        }
        const float bv = bias[col];
#pragma unroll
        for (int v = 0; v < 8; ++v) out[nt][v] = acc[v] + bv;
    }

    // layernorm over each 128-wide row: per-lane partial over 8 N-tiles,
    // then 16-lane butterfly (xor masks <16 stay inside each half).
#pragma unroll
    for (int v = 0; v < 8; ++v) {
        float s = 0.0f;
#pragma unroll
        for (int nt = 0; nt < 8; ++nt) s += out[nt][v];
        s += __shfl_xor(s, 1); s += __shfl_xor(s, 2);
        s += __shfl_xor(s, 4); s += __shfl_xor(s, 8);
        const float mean = s * (1.0f / 128.0f);
        float q2 = 0.0f;
#pragma unroll
        for (int nt = 0; nt < 8; ++nt) {
            float d = out[nt][v] - mean;
            q2 += d * d;
        }
        q2 += __shfl_xor(q2, 1); q2 += __shfl_xor(q2, 2);
        q2 += __shfl_xor(q2, 4); q2 += __shfl_xor(q2, 8);
        const float rstd = rsqrtf(q2 * (1.0f / 128.0f) + 1e-5f);
#pragma unroll
        for (int nt = 0; nt < 8; ++nt) {
            const int col = nt * 16 + r;
            out[nt][v] = (out[nt][v] - mean) * rstd * g[col] + be[col];
        }
    }
}

// 5-neighbor gather-sum (adj @ h), element-parallel over all 160*128 elements.
__device__ __forceinline__ void aggregate(const __bf16* __restrict__ src,
                                          __bf16* __restrict__ dst,
                                          const int* __restrict__ nbr, int tid) {
    for (int e = tid; e < ROWS * HID; e += NTHREADS) {
        const int row = e >> 7, d = e & 127;
        const int* nb = &nbr[row * 5];
        float s = 0.0f;
#pragma unroll
        for (int k = 0; k < 5; ++k) s += (float)src[nb[k] * STR + d];
        dst[row * STR + d] = (__bf16)s;
    }
}

__global__ __launch_bounds__(NTHREADS)
void hbond_gnn_kernel(const float* __restrict__ x,
                      const float* __restrict__ wE, const float* __restrict__ bE,
                      const float* __restrict__ w1, const float* __restrict__ b1,
                      const float* __restrict__ w2, const float* __restrict__ b2,
                      const float* __restrict__ g1, const float* __restrict__ be1,
                      const float* __restrict__ g2, const float* __restrict__ be2,
                      float* __restrict__ out) {
    extern __shared__ char smem[];
    // Single weight buffer: holds w1^T for GEMM1, re-staged with w2^T for GEMM2.
    // Total 139,008 B < 160 KB -> 2 workgroups per 320 KB WGP (10 waves).
    __bf16* sWt  = (__bf16*)(smem);            // 128*136*2 = 34816
    __bf16* sH   = (__bf16*)(smem + 34816);    // 160*136*2 = 43520
    __bf16* sHa  = (__bf16*)(smem + 78336);    // 43520
    float*  sWe  = (float*)(smem + 121856);    // 9*128*4 = 4608
    float*  sX   = (float*)(smem + 126464);    // 8*20*9*4 = 5760
    float*  sB   = (float*)(smem + 132224);    // 7*128*4 = 3584 [bE,b1,g1,be1,b2,g2,be2]
    int*    sNbr = (int*)(smem + 135808);      // 160*5*4 = 3200 -> total 139008 B

    const int tid  = threadIdx.x;
    const int lane = tid & 31;
    const int wave = tid >> 5;
    const long base = (long)blockIdx.x * BPW;

    // ---- async staging of contiguous f32 blobs via Tensor Data Mover ----
    if (tid == 0)
        tdm_copy_dwords(x + base * 180, (unsigned)(unsigned long long)(void*)sX,
                        BPW * 20 * 9);                       // 1440 dwords
    if (tid == 32)
        tdm_copy_dwords(wE, (unsigned)(unsigned long long)(void*)sWe,
                        9 * HID);                            // 1152 dwords

    // ---- stage w1 transposed+bf16 (VALU: TDM can't convert/transpose) ----
    for (int i = tid; i < HID * HID; i += NTHREADS) {
        const int k = i >> 7, n = i & 127;
        sWt[n * STR + k] = (__bf16)w1[i];
    }
    if (tid < HID) {
        sB[tid]       = bE[tid];
        sB[128 + tid] = b1[tid];
        sB[256 + tid] = g1[tid];
        sB[384 + tid] = be1[tid];
        sB[512 + tid] = b2[tid];
        sB[640 + tid] = g2[tid];
        sB[768 + tid] = be2[tid];
    }
    if (wave < 2) __builtin_amdgcn_s_wait_tensorcnt(0);      // TDM done before barrier
    __syncthreads();

    // ---- 5-NN per node (one thread per row; ties -> lowest index like top_k) ----
    {
        const int b = tid / 20, i = tid % 20;
        const float* pb = &sX[b * 180];
        const float px = pb[i * 9 + 6], py = pb[i * 9 + 7], pz = pb[i * 9 + 8];
        float d2[20];
#pragma unroll
        for (int j = 0; j < 20; ++j) {
            const float dx = pb[j * 9 + 6] - px;
            const float dy = pb[j * 9 + 7] - py;
            const float dz = pb[j * 9 + 8] - pz;
            d2[j] = dx * dx + dy * dy + dz * dz;
        }
#pragma unroll
        for (int k = 0; k < 5; ++k) {
            float best = 3.4e38f; int bj = 0;
#pragma unroll
            for (int j = 0; j < 20; ++j)
                if (d2[j] < best) { best = d2[j]; bj = j; }
            d2[bj] = 3.4e38f;
            sNbr[tid * 5 + k] = b * 20 + bj;
        }
    }

    // ---- embed: h0 = x @ wE + bE  (tiny K=9, VALU) -> sH (bf16) ----
    for (int e = tid; e < ROWS * HID; e += NTHREADS) {
        const int row = e >> 7, d = e & 127;
        float s = sB[d];
        const float* xr = &sX[row * 9];
#pragma unroll
        for (int k = 0; k < 9; ++k) s += xr[k] * sWe[k * HID + d];
        sH[row * STR + d] = (__bf16)s;
    }
    __syncthreads();

    // ---- h = adj @ h0 ----
    aggregate(sH, sHa, sNbr, tid);
    __syncthreads();

    // ---- h = gelu(LN(h @ w1 + b1))  [WMMA]  -> sH ----
    for (int t = wave; t < 10; t += 5) {
        float o[8][8];
        mm_ln(sHa, sWt, sB + 128, sB + 256, sB + 384, t, lane, o);
        const int r = lane & 15, half = lane >> 4;
#pragma unroll
        for (int nt = 0; nt < 8; ++nt)
#pragma unroll
            for (int v = 0; v < 8; ++v)
                sH[(t * 16 + v + half * 8) * STR + nt * 16 + r] =
                    (__bf16)gelu_exact(o[nt][v]);
    }
    __syncthreads();

    // ---- re-stage weight buffer with w2^T; h2 = adj @ h ----
    for (int i = tid; i < HID * HID; i += NTHREADS) {
        const int k = i >> 7, n = i & 127;
        sWt[n * STR + k] = (__bf16)w2[i];
    }
    aggregate(sH, sHa, sNbr, tid);
    __syncthreads();

    // ---- h = gelu(h + LN(h2 @ w2 + b2))  [WMMA]  -> sHa ----
    for (int t = wave; t < 10; t += 5) {
        float o[8][8];
        mm_ln(sHa, sWt, sB + 512, sB + 640, sB + 768, t, lane, o);
        const int r = lane & 15, half = lane >> 4;
#pragma unroll
        for (int nt = 0; nt < 8; ++nt)
#pragma unroll
            for (int v = 0; v < 8; ++v) {
                const int idx = (t * 16 + v + half * 8) * STR + nt * 16 + r;
                const float h = (float)sH[idx] + o[nt][v];
                sHa[idx] = (__bf16)gelu_exact(h);  // own-tile rows only: race-free
            }
    }
    __syncthreads();

    // ---- out[b][d] = max over 20 nodes ----
    for (int e = tid; e < BPW * HID; e += NTHREADS) {
        const int b = e >> 7, d = e & 127;
        float m = -3.4e38f;
#pragma unroll
        for (int n = 0; n < 20; ++n)
            m = fmaxf(m, (float)sHa[(b * 20 + n) * STR + d]);
        out[(base + b) * HID + d] = m;
    }
}

extern "C" void kernel_launch(void* const* d_in, const int* in_sizes, int n_in,
                              void* d_out, int out_size, void* d_ws, size_t ws_size,
                              hipStream_t stream) {
    const float* x   = (const float*)d_in[0];
    const float* wE  = (const float*)d_in[1];
    const float* bE  = (const float*)d_in[2];
    const float* w1  = (const float*)d_in[3];
    const float* b1  = (const float*)d_in[4];
    const float* w2  = (const float*)d_in[5];
    const float* b2  = (const float*)d_in[6];
    const float* g1  = (const float*)d_in[7];
    const float* be1 = (const float*)d_in[8];
    const float* g2  = (const float*)d_in[9];
    const float* be2 = (const float*)d_in[10];
    float* out = (float*)d_out;

    const int B = in_sizes[0] / (20 * 9);        // 32768
    const int nblocks = B / BPW;                 // 4096
    const size_t shmem = 139008;                 // < 160 KB -> 2 WGs per WGP

    (void)hipFuncSetAttribute((const void*)hbond_gnn_kernel,
                              hipFuncAttributeMaxDynamicSharedMemorySize,
                              (int)shmem);

    hbond_gnn_kernel<<<nblocks, NTHREADS, shmem, stream>>>(
        x, wE, bE, w1, b1, w2, b2, g1, be1, g2, be2, out);
}
